// BasicModel_4191888081018
// MI455X (gfx1250) — matrix operand order repaired and checked by reading, not compiled, not run on hardware
//
#include <hip/hip_runtime.h>
#include <hip/hip_bf16.h>
#include <math.h>

// MPNN forward for MI455X (gfx1250, wave32, WMMA).
// - Factorized edge-message einsum: 19.3 GF / 604 MB -> 0.78 GF / <13 MB (L2).
// - All GEMMs: v_wmma_f32_16x16x32_f16 (f16 operands, f32 accumulate).
// - All f16 B operands stored in fragment-swizzled layout (lane*16 halves
//   contiguous) so each B fragment is one 32B load (2x b128) instead of 16
//   strided u16 loads. A fragments load as 2x contiguous v8h (b128).
// - K4: persistent 1024-thread workgroup; node_state in WGP LDS; GRU weights
//   staged global->LDS via TDM tensor_load_to_lds (s_wait_tensorcnt) when the
//   builtin exists, with a plain-load fallback.

typedef __attribute__((ext_vector_type(16))) _Float16 v16h;
typedef __attribute__((ext_vector_type(8)))  _Float16 v8h;
typedef __attribute__((ext_vector_type(8)))  float    v8f;
typedef __attribute__((ext_vector_type(4)))  unsigned int u32x4;
typedef __attribute__((ext_vector_type(8)))  int      i32x8;
typedef __attribute__((ext_vector_type(4)))  int      i32x4;

#define B_    16
#define N_    48
#define FN_   64
#define FE_   16
#define FM_   64
#define OUT_  128
#define STEPS_ 6
#define EPS_  1e-5f
#define ROWS  (B_ * N_)        /* 768 */
#define PAIRS (B_ * N_ * N_)   /* 36864 */

#if __has_builtin(__builtin_amdgcn_tensor_load_to_lds)
#define HAVE_TDM 1
#endif

// ---------------------------------------------------------------------------
// WMMA fragment helpers per CDNA5 ISA §7.12.2 (wave32 layouts).
// A (16x32 f16): lane<16 -> row=lane, K-base 0; lane>=16 -> row=lane-16, K-base 16 (contiguous k order, as for B).
//   Per lane this is two contiguous 8-half runs: [kb..kb+7], [kb+8..kb+15].
// B (32x16 f16): lane<16 holds col (lane&15) K 0..15; lane>=16 K 16..31.
//   Swizzled storage: element i of lane l lives at tile[l*16 + i] so a whole
//   fragment is one aligned 32-byte per-lane load.
// C/D (16x16 f32, 8 VGPRs): col = lane&15, row = r + (lane<16 ? 0 : 8).
// ---------------------------------------------------------------------------
static __device__ inline int lane_id() { return threadIdx.x & 31; }

static __device__ inline v16h frag_a_f16(const _Float16* s, int lda) {
  int l = lane_id(), mr = l & 15, kb = (l < 16) ? 0 : 16;
  const _Float16* p = s + mr * lda + kb;
  v8h lo = *(const v8h*)(p);        // K kb..kb+7    -> VGPRs 0-3
  v8h hi = *(const v8h*)(p + 8);    // K kb+8..kb+15 -> VGPRs 4-7
  v16h a;
#pragma unroll
  for (int i = 0; i < 8; ++i) { a[i] = lo[i]; a[8 + i] = hi[i]; }
  return a;
}

static __device__ inline v16h frag_a_f32pad(const float* s, int lda, int kmax) {
  int l = lane_id(), mr = l & 15, kb = (l < 16) ? 0 : 16;
  v16h a;
#pragma unroll
  for (int vv = 0; vv < 8; ++vv) {
    int k = ((vv & 3) << 1) + ((vv >> 2) << 3) + kb;
    a[2 * vv]     = (k     < kmax) ? (_Float16)s[mr * lda + k]     : (_Float16)0.f;
    a[2 * vv + 1] = (k + 1 < kmax) ? (_Float16)s[mr * lda + k + 1] : (_Float16)0.f;
  }
  return a;
}

// One 32x16 B tile in swizzled storage occupies 512 halves (1 KB).
static __device__ inline v16h frag_b_sw(const _Float16* tile) {
  return *(const v16h*)(tile + lane_id() * 16);
}

static __device__ inline v8f wmma32(v16h a, v16h b, v8f c) {
  return __builtin_amdgcn_wmma_f32_16x16x32_f16(false, a, false, b, (short)0, c,
                                                false, false);
}

static __device__ inline void zero8(v8f& c) {
#pragma unroll
  for (int i = 0; i < 8; ++i) c[i] = 0.f;
}

static __device__ inline void store_c_f16(_Float16* d, int ldc, v8f c) {
  int l = lane_id(), nc = l & 15, mb = (l < 16) ? 0 : 8;
#pragma unroll
  for (int r = 0; r < 8; ++r) d[(mb + r) * ldc + nc] = (_Float16)c[r];
}

static __device__ inline float sigf(float x) { return 1.f / (1.f + __expf(-x)); }

// ---------------------------------------------------------------------------
// K0: f16 conversions into fragment-swizzled layouts + per-batch col-sums of
//     afm (bias term of the factorized message einsum).
// Swizzle inverse map: flat i -> (tile, lane=(i>>4)&31, ii=i&15);
//   k = ktile*32 + (lane&16 ? 16 : 0) + ii ; col = ctile*16 + (lane&15)
// ---------------------------------------------------------------------------
__global__ void k0_convert(const float* __restrict__ W_e2,
                           const float* __restrict__ afm,
                           const float* __restrict__ W_i,
                           const float* __restrict__ W_j,
                           _Float16* __restrict__ w2h,
                           _Float16* __restrict__ afmh,
                           _Float16* __restrict__ wi_h,
                           _Float16* __restrict__ wj_h,
                           float* __restrict__ asum) {
  int tid = blockIdx.x * blockDim.x + threadIdx.x;
  int nth = gridDim.x * blockDim.x;
  // W_e2 as 64 per-f B matrices (64x64), tiles [kh(2)][ct(4)] of 512 halves
  for (int i = tid; i < FM_ * FN_ * FM_; i += nth) {
    int f = i >> 12, r = i & 4095;
    int tile = r >> 9, khx = tile >> 2, ctx = tile & 3;
    int ln = (r >> 4) & 31, ii = r & 15;
    int k = khx * 32 + ((ln & 16) ? 16 : 0) + ii;
    int col = ctx * 16 + (ln & 15);
    w2h[i] = (_Float16)W_e2[f * 4096 + k * 64 + col];
  }
  // afm f16 stays row-major (A operand)
  for (int i = tid; i < ROWS * FN_; i += nth) afmh[i] = (_Float16)afm[i];
  // W_i / W_j (128x128) B tiles [kh(4)][ct(8)]
  for (int i = tid; i < 2 * FN_ * OUT_; i += nth) {
    int tile = i >> 9, khx = tile >> 3, ctx = tile & 7;
    int ln = (i >> 4) & 31, ii = i & 15;
    int k = khx * 32 + ((ln & 16) ? 16 : 0) + ii;
    int col = ctx * 16 + (ln & 15);
    wi_h[i] = (_Float16)W_i[k * 128 + col];
    wj_h[i] = (_Float16)W_j[k * 128 + col];
  }
  for (int i = tid; i < B_ * FN_; i += nth) {
    int b = i >> 6, c = i & 63;
    float s = 0.f;
    for (int j = 0; j < N_; ++j) s += afm[(b * N_ + j) * FN_ + c];
    asum[i] = s;
  }
}

// ---------------------------------------------------------------------------
// K1: h_e = relu(bfm @ W_e1 + b_e1) -- (36864 x 16) @ (16 x 64), K padded to 32.
// ---------------------------------------------------------------------------
__global__ __launch_bounds__(128) void k1_edge_hidden(
    const float* __restrict__ bfm, const float* __restrict__ W_e1,
    const float* __restrict__ b_e1, _Float16* __restrict__ h_e) {
  __shared__ __align__(32) _Float16 sB[4 * 512];   // swizzled, K rows >=16 zero
  int tid = threadIdx.x;
  for (int i = tid; i < 4 * 512; i += 128) {
    int ctx = i >> 9, ln = (i >> 4) & 31, ii = i & 15;
    int k = ((ln & 16) ? 16 : 0) + ii;
    int col = ctx * 16 + (ln & 15);
    sB[i] = (k < FE_) ? (_Float16)W_e1[k * 64 + col] : (_Float16)0.f;
  }
  __syncthreads();

  int wave = tid >> 5;
  int tile = blockIdx.x * 4 + wave;            // 2304 row tiles of 16
  const float* arow = bfm + tile * 16 * FE_;   // lda = 16
  v16h a = frag_a_f32pad(arow, FE_, FE_);

  int l = lane_id(), nc = l & 15;
#pragma unroll
  for (int ct = 0; ct < 4; ++ct) {
    v8f c; zero8(c);
    c = wmma32(a, frag_b_sw(sB + ct * 512), c);
    float bias = b_e1[ct * 16 + nc];
#pragma unroll
    for (int r = 0; r < 8; ++r) {
      float x = c[r] + bias;
      c[r] = x > 0.f ? x : 0.f;
    }
    store_c_f16(h_e + tile * 16 * 64 + ct * 16, 64, c);
  }
}

// ---------------------------------------------------------------------------
// K2: T[b,j,f,m] = sum_n afm[b,j,n] * W_e2[f, n*64+m]
//     64 per-f GEMMs of (768 x 64) @ (64 x 64); one (f, rowtile) per wave.
//     Output scatter-stored directly in K3's swizzled-B layout:
//       Tsw[b][ks(96)][ct(4)][lane][i], k-row = j*64+f, col = m.
// ---------------------------------------------------------------------------
__global__ __launch_bounds__(256) void k2_factor(
    const _Float16* __restrict__ afmh, const _Float16* __restrict__ w2h,
    _Float16* __restrict__ Tsw) {
  int wave = threadIdx.x >> 5;
  int task = blockIdx.x * 8 + wave;   // 64 * 48 = 3072 tasks
  int f = task / 48, rt = task % 48;
  const _Float16* A0 = afmh + rt * 16 * FN_;   // lda 64
  const _Float16* Bb = w2h + f * FN_ * FM_;    // swizzled tiles [kh(2)][ct(4)]

  v8f acc[4];
#pragma unroll
  for (int i = 0; i < 4; ++i) zero8(acc[i]);
#pragma unroll
  for (int kh = 0; kh < 2; ++kh) {
    v16h a = frag_a_f16(A0 + kh * 32, FN_);
#pragma unroll
    for (int ct = 0; ct < 4; ++ct)
      acc[ct] = wmma32(a, frag_b_sw(Bb + (kh * 4 + ct) * 512), acc[ct]);
  }
  // scatter into swizzled T: k-row = j*64 + f -> ks = 2j + (f>>5),
  // lane_t = (f&16 ? 16 : 0) + col%16, i = f&15
  int l = lane_id(), nc = l & 15, mb = (l < 16) ? 0 : 8;
  int ks_hi = f >> 5;
  int lane_t = ((f & 16) ? 16 : 0) + nc;
  int ii = f & 15;
#pragma unroll
  for (int ct = 0; ct < 4; ++ct) {
#pragma unroll
    for (int r = 0; r < 8; ++r) {
      int bj = rt * 16 + mb + r;
      int b = bj / 48, j = bj % 48;   // 16-row tiles never cross a batch (48=3*16)
      Tsw[(size_t)b * 196608 + (size_t)(2 * j + ks_hi) * 2048 + ct * 512 +
          lane_t * 16 + ii] = (_Float16)acc[ct][r];
    }
  }
}

// ---------------------------------------------------------------------------
// K3: m[b,i,m] = sum_{jf} h_e[b,i,(jf)] * Tsw[b,(jf),m] + sum_n b_e2[n,m]*asum[b,n]
//     Per batch: (48 x 3072) @ (3072 x 64), K-deep WMMA loop (96 iterations).
// ---------------------------------------------------------------------------
__global__ __launch_bounds__(256) void k3_messages(
    const _Float16* __restrict__ h_e, const _Float16* __restrict__ Tsw,
    const float* __restrict__ b_e2, const float* __restrict__ asum,
    float* __restrict__ m) {
  int wave = threadIdx.x >> 5;
  int task = blockIdx.x * 8 + wave;   // 48 tasks
  int b = task / 3, it = task % 3;
  const _Float16* A0 = h_e + (size_t)(b * N_ + it * 16) * 3072;  // lda 3072
  const _Float16* B0 = Tsw + (size_t)b * 196608;                 // swizzled

  v8f acc[4];
#pragma unroll
  for (int i = 0; i < 4; ++i) zero8(acc[i]);
  for (int ks = 0; ks < 96; ++ks) {              // 96 * 32 = 3072
    if ((ks & 7) == 0) __builtin_prefetch(A0 + (ks + 8) * 32, 0, 0);
    v16h a = frag_a_f16(A0 + ks * 32, 3072);
#pragma unroll
    for (int ct = 0; ct < 4; ++ct)
      acc[ct] = wmma32(a, frag_b_sw(B0 + ks * 2048 + ct * 512), acc[ct]);
  }

  int l = lane_id(), nc = l & 15, mb = (l < 16) ? 0 : 8;
  float* dst = m + (b * N_ + it * 16) * 64;
#pragma unroll
  for (int ct = 0; ct < 4; ++ct) {
    int col = ct * 16 + nc;
    float bias = 0.f;
    for (int q = 0; q < FN_; ++q) bias += b_e2[q * 64 + col] * asum[b * 64 + q];
#pragma unroll
    for (int r = 0; r < 8; ++r) dst[(mb + r) * 64 + col] = acc[ct][r] + bias;
  }
}

// ---------------------------------------------------------------------------
// K4: persistent tail. One 1024-thread workgroup (32 waves on one WGP).
//   - TDM-stage W_ih/W_hh (f32) global->LDS, swizzle-convert to f16
//   - masked BN stats of m; gi = bn(m) @ W_ih + b_ih  (step-invariant)
//   - 6x: gh = h @ W_hh (WMMA, h in LDS), GRU pointwise, masked BN of upd
//   - gated readout: sig(x@W_i+b_i)*(x@W_j+b_j), per-batch row sum
// LDS: h (768x64 f32, 192KB) + swizzled W f16 + reductions (~249KB of 320KB).
// ---------------------------------------------------------------------------
#define K4_SMEM 255040
__global__ __launch_bounds__(1024) void k4_tail(
    const float* __restrict__ afm, const float* __restrict__ mask,
    const float* __restrict__ W_ih, const float* __restrict__ W_hh,
    const float* __restrict__ b_ih, const float* __restrict__ b_hh,
    const float* __restrict__ g_ma, const float* __restrict__ bt_ma,
    const float* __restrict__ g_bn, const float* __restrict__ bt_bn,
    const _Float16* __restrict__ wi_h, const _Float16* __restrict__ wj_h,
    const float* __restrict__ b_i, const float* __restrict__ b_j,
    const float* __restrict__ m, float* __restrict__ gi,
    float* __restrict__ out) {
  extern __shared__ char smem[];
  float*    h     = (float*)smem;                           // 768*64 f32 (also TDM staging)
  _Float16* wih   = (_Float16*)(smem + 196608);             // swizzled [kh2][ct12]*512
  _Float16* whh   = (_Float16*)(smem + 196608 + 24576);     // swizzled [kh2][ct12]*512
  float*    red   = (float*)(smem + 245760);                // 64
  float*    red2  = red + 64;                               // 64
  float*    meanv = red + 128;                              // 64
  float*    invv  = red + 192;                              // 64
  float*    ldsOut= red + 256;                              // 16*128
  float*    cntp  = ldsOut + 2048;

  int tid = threadIdx.x, l = tid & 31, wave = tid >> 5;
  int nc = l & 15, mb = (l < 16) ? 0 : 8, am = l & 15, kb = (l < 16) ? 0 : 16;

  // ---- stage 0: zero reducers ----
  if (tid == 0) *cntp = 0.f;
  for (int i = tid; i < 64; i += 1024) { red[i] = 0.f; red2[i] = 0.f; }
  for (int i = tid; i < B_ * OUT_; i += 1024) ldsOut[i] = 0.f;

  // ---- stage 1: GRU weights global -> LDS via Tensor Data Mover ----
#if defined(HAVE_TDM)
  if (wave == 0) {
    // D# for a 1-D tile of 12288 f32 elements (48 KB): group0 + group1.
    unsigned long long ga = (unsigned long long)(size_t)W_ih;
    unsigned lds0 = (unsigned)(size_t)(void*)h;
    u32x4 g0 = {0u, 0u, 0u, 0u};
    i32x8 g1 = {0, 0, 0, 0, 0, 0, 0, 0};
    i32x4 z4 = {0, 0, 0, 0};
    i32x8 z8 = {0, 0, 0, 0, 0, 0, 0, 0};
    g0[0] = 1u;                                           // count=1 (user mode)
    g0[1] = lds0;                                         // lds_addr
    g0[2] = (unsigned)(ga & 0xffffffffu);                 // global_addr[31:0]
    g0[3] = (unsigned)((ga >> 32) & 0x1ffffffu) | (2u << 30); // addr[56:32]|type=2
    g1[0] = (2 << 16);                                    // data_size = 4 B
    g1[1] = (int)(12288u << 16);                          // tensor_dim0[15:0]
    g1[2] = (1 << 16);                                    // dim0 hi=0, tensor_dim1=1
    g1[3] = (int)(12288u << 16);                          // tile_dim0 = 12288
    g1[5] = 12288;                                        // tensor_dim0_stride
#if __clang_major__ >= 23
    __builtin_amdgcn_tensor_load_to_lds(g0, g1, z4, z4, z8, 0);
#else
    __builtin_amdgcn_tensor_load_to_lds(g0, g1, z4, z4, 0);
#endif
    ga = (unsigned long long)(size_t)W_hh;
    g0[1] = lds0 + 49152u;
    g0[2] = (unsigned)(ga & 0xffffffffu);
    g0[3] = (unsigned)((ga >> 32) & 0x1ffffffu) | (2u << 30);
#if __clang_major__ >= 23
    __builtin_amdgcn_tensor_load_to_lds(g0, g1, z4, z4, z8, 0);
#else
    __builtin_amdgcn_tensor_load_to_lds(g0, g1, z4, z4, 0);
#endif
    __builtin_amdgcn_s_wait_tensorcnt(0);
  }
  __syncthreads();
  const float* sWih = (const float*)h;
  const float* sWhh = ((const float*)h) + 12288;
#else
  __syncthreads();
  const float* sWih = W_ih;
  const float* sWhh = W_hh;
#endif
  // swizzle-convert to f16 B tiles [kh(2)][ct(12)]
  for (int i = tid; i < 2 * 12 * 512; i += 1024) {
    int tile = i >> 9, khx = tile / 12, ctx = tile % 12;
    int ln = (i >> 4) & 31, ii = i & 15;
    int k = khx * 32 + ((ln & 16) ? 16 : 0) + ii;
    int col = ctx * 16 + (ln & 15);
    wih[i] = (_Float16)sWih[k * 192 + col];
    whh[i] = (_Float16)sWhh[k * 192 + col];
  }
  __syncthreads();

  // ---- stage 2: h = afm; masked BN stats of m ----
  for (int i = tid; i < ROWS * FN_; i += 1024) {
    h[i] = afm[i];
    int row = i >> 6, c = i & 63;
    float x = m[i] * mask[row];
    atomicAdd(&red[c], x);
    atomicAdd(&red2[c], x * x);
  }
  for (int i = tid; i < ROWS; i += 1024) atomicAdd(cntp, mask[i]);
  __syncthreads();
  if (tid < 64) {
    float cnt = *cntp;
    float mu = red[tid] / cnt;
    meanv[tid] = mu;
    invv[tid] = rsqrtf(red2[tid] / cnt - mu * mu + EPS_);
  }
  __syncthreads();

  // ---- stage 3: gi = bn(m) @ W_ih + b_ih : (768x64)@(64x192), A normalized on load
  for (int t = wave; t < 48; t += 32) {
    v8f acc[12];
#pragma unroll
    for (int i = 0; i < 12; ++i) zero8(acc[i]);
#pragma unroll
    for (int kh = 0; kh < 2; ++kh) {
      v16h a;
      int row = t * 16 + am;
      float mk = mask[row];
#pragma unroll
      for (int vv = 0; vv < 8; ++vv) {
        int k0 = ((vv & 3) << 1) + ((vv >> 2) << 3) + kb + kh * 32;
#pragma unroll
        for (int p = 0; p < 2; ++p) {
          int k = k0 + p;
          float x = m[row * 64 + k];
          x = ((x - meanv[k]) * invv[k] * g_ma[k] + bt_ma[k]) * mk;
          a[2 * vv + p] = (_Float16)x;
        }
      }
#pragma unroll
      for (int ct = 0; ct < 12; ++ct)
        acc[ct] = wmma32(a, frag_b_sw(wih + (kh * 12 + ct) * 512), acc[ct]);
    }
    float* dst = gi + t * 16 * 192;
#pragma unroll
    for (int ct = 0; ct < 12; ++ct) {
      int col = ct * 16 + nc;
      float bias = b_ih[col];
#pragma unroll
      for (int r = 0; r < 8; ++r) dst[(mb + r) * 192 + col] = acc[ct][r] + bias;
    }
  }
  __threadfence();   // gi lives in global; make visible to all waves in WG
  __syncthreads();

  // ---- stage 4: 6 GRU steps ----
  for (int step = 0; step < STEPS_; ++step) {
    if (tid < 64) { red[tid] = 0.f; red2[tid] = 0.f; }
    __syncthreads();
    for (int t = wave; t < 48; t += 32) {
      v8f acc[12];
#pragma unroll
      for (int i = 0; i < 12; ++i) zero8(acc[i]);
#pragma unroll
      for (int kh = 0; kh < 2; ++kh) {
        v16h a;
#pragma unroll
        for (int vv = 0; vv < 8; ++vv) {
          int k = ((vv & 3) << 1) + ((vv >> 2) << 3) + kb + kh * 32;
          a[2 * vv]     = (_Float16)h[(t * 16 + am) * 64 + k];
          a[2 * vv + 1] = (_Float16)h[(t * 16 + am) * 64 + k + 1];
        }
#pragma unroll
        for (int ct = 0; ct < 12; ++ct)
          acc[ct] = wmma32(a, frag_b_sw(whh + (kh * 12 + ct) * 512), acc[ct]);
      }
#pragma unroll
      for (int ct = 0; ct < 4; ++ct) {
        int q = ct * 16 + nc;
#pragma unroll
        for (int r = 0; r < 8; ++r) {
          int row = t * 16 + mb + r;
          float mk = mask[row];
          const float* gir = gi + row * 192;
          float rg = sigf(gir[q]        + acc[ct][r]     + b_hh[q]);
          float zg = sigf(gir[64 + q]   + acc[ct + 4][r] + b_hh[64 + q]);
          float ng = tanhf(gir[128 + q] + rg * (acc[ct + 8][r] + b_hh[128 + q]));
          float hv = h[row * 64 + q];
          float upd = ((1.f - zg) * ng + zg * hv) * mk;
          h[row * 64 + q] = upd;
          atomicAdd(&red[q], upd);
          atomicAdd(&red2[q], upd * upd);
        }
      }
    }
    __syncthreads();
    if (tid < 64) {
      float cnt = *cntp;
      float mu = red[tid] / cnt;
      meanv[tid] = mu;
      invv[tid] = rsqrtf(red2[tid] / cnt - mu * mu + EPS_);
    }
    __syncthreads();
    for (int i = tid; i < ROWS * FN_; i += 1024) {
      int row = i >> 6, c = i & 63;
      h[i] = ((h[i] - meanv[c]) * invv[c] * g_bn[c] + bt_bn[c]) * mask[row];
    }
    __syncthreads();
  }

  // ---- stage 5: gated readout x=[h,afm]; sig(x@Wi+bi)*(x@Wj+bj); batch row-sum
  for (int t = wave; t < 48; t += 32) {
    int bidx = t / 3;
    v16h afr[4];
#pragma unroll
    for (int kh = 0; kh < 4; ++kh) {
#pragma unroll
      for (int vv = 0; vv < 8; ++vv) {
        int k = ((vv & 3) << 1) + ((vv >> 2) << 3) + kb + kh * 32;
        int row = t * 16 + am;
        float x0, x1;
        if (kh < 2) { x0 = h[row * 64 + k];          x1 = h[row * 64 + k + 1]; }
        else        { x0 = afm[row * 64 + (k - 64)]; x1 = afm[row * 64 + (k - 63)]; }
        afr[kh][2 * vv]     = (_Float16)x0;
        afr[kh][2 * vv + 1] = (_Float16)x1;
      }
    }
#pragma unroll
    for (int ct = 0; ct < 8; ++ct) {
      v8f ai, aj; zero8(ai); zero8(aj);
#pragma unroll
      for (int kh = 0; kh < 4; ++kh) {
        ai = wmma32(afr[kh], frag_b_sw(wi_h + (kh * 8 + ct) * 512), ai);
        aj = wmma32(afr[kh], frag_b_sw(wj_h + (kh * 8 + ct) * 512), aj);
      }
      int col = ct * 16 + nc;
#pragma unroll
      for (int r = 0; r < 8; ++r) {
        int row = t * 16 + mb + r;
        float g = sigf(ai[r] + b_i[col]) * (aj[r] + b_j[col]) * mask[row];
        atomicAdd(&ldsOut[bidx * OUT_ + col], g);
      }
    }
  }
  __syncthreads();
  for (int i = tid; i < B_ * OUT_; i += 1024) out[i] = ldsOut[i];
}

// ---------------------------------------------------------------------------
// Workspace layout (bytes; ~12.5 MB total, L2-resident on MI455X's 192 MB L2)
// ---------------------------------------------------------------------------
#define O_W2H   0u         /* 64*64*64 f16 swizzled = 524288  */
#define O_AFMH  524288u    /* 768*64 f16 row-major  = 98304   */
#define O_WIH16 622592u    /* 128*128 f16 swizzled  = 32768   */
#define O_WJH16 655360u    /* 128*128 f16 swizzled  = 32768   */
#define O_ASUM  688128u    /* 16*64 f32             = 4096    */
#define O_HE    692224u    /* 36864*64 f16          = 4718592 */
#define O_T     5410816u   /* 768*4096 f16 swizzled = 6291456 */
#define O_M     11702272u  /* 768*64 f32            = 196608  */
#define O_GI    11898880u  /* 768*192 f32           = 589824  */

extern "C" void kernel_launch(void* const* d_in, const int* in_sizes, int n_in,
                              void* d_out, int out_size, void* d_ws, size_t ws_size,
                              hipStream_t stream) {
  (void)in_sizes; (void)n_in; (void)out_size; (void)ws_size;
  const float* afm   = (const float*)d_in[0];
  const float* bfm   = (const float*)d_in[1];
  /* d_in[2] = adj: unused by the reference forward */
  const float* mask  = (const float*)d_in[3];
  const float* W_e1  = (const float*)d_in[4];
  const float* b_e1  = (const float*)d_in[5];
  const float* W_e2  = (const float*)d_in[6];
  const float* b_e2  = (const float*)d_in[7];
  const float* W_ih  = (const float*)d_in[8];
  const float* W_hh  = (const float*)d_in[9];
  const float* b_ih  = (const float*)d_in[10];
  const float* b_hh  = (const float*)d_in[11];
  const float* g_ma  = (const float*)d_in[12];
  const float* bt_ma = (const float*)d_in[13];
  const float* g_bn  = (const float*)d_in[14];
  const float* bt_bn = (const float*)d_in[15];
  const float* W_i   = (const float*)d_in[16];
  const float* b_i   = (const float*)d_in[17];
  const float* W_j   = (const float*)d_in[18];
  const float* b_j   = (const float*)d_in[19];

  char* ws = (char*)d_ws;
  _Float16* w2h  = (_Float16*)(ws + O_W2H);
  _Float16* afmh = (_Float16*)(ws + O_AFMH);
  _Float16* wi_h = (_Float16*)(ws + O_WIH16);
  _Float16* wj_h = (_Float16*)(ws + O_WJH16);
  float*    asum = (float*)   (ws + O_ASUM);
  _Float16* h_e  = (_Float16*)(ws + O_HE);
  _Float16* Tsw  = (_Float16*)(ws + O_T);
  float*    mbuf = (float*)   (ws + O_M);
  float*    gibf = (float*)   (ws + O_GI);

  k0_convert<<<256, 256, 0, stream>>>(W_e2, afm, W_i, W_j, w2h, afmh, wi_h, wj_h, asum);
  k1_edge_hidden<<<PAIRS / (16 * 4), 128, 0, stream>>>(bfm, W_e1, b_e1, h_e);
  k2_factor<<<(64 * 48) / 8, 256, 0, stream>>>(afmh, w2h, Tsw);
  k3_messages<<<48 / 8, 256, 0, stream>>>(h_e, Tsw, b_e2, asum, mbuf);
  k4_tail<<<1, 1024, K4_SMEM, stream>>>(afm, mask, W_ih, W_hh, b_ih, b_hh,
                                        g_ma, bt_ma, g_bn, bt_bn,
                                        wi_h, wj_h, b_i, b_j,
                                        mbuf, gibf, (float*)d_out);
}
